// VQEmbeddingMovingAverage_54331336294546
// MI455X (gfx1250) — matrix operand order repaired and checked by reading, not compiled, not run on hardware
//
#include <hip/hip_runtime.h>
#include <hip/hip_bf16.h>
#include <stdint.h>

// Problem constants (from reference): B=16, T=2048, D=256, K=1024
#define N_ROWS   32768      // B*T
#define DDIM     256
#define KCODES   1024
#define TILE_N   16         // codes per WMMA tile
#define NTILES   (KCODES / TILE_N)   // 64
#define KCHUNK   32         // K-dim per bf16 WMMA
#define NKCH     (DDIM / KCHUNK)     // 8
#define LDS_STRIDE 264      // 256 bf16 + 8 pad (528B row stride -> conflict-free)

typedef __attribute__((ext_vector_type(16))) __bf16 v16bf;
typedef __attribute__((ext_vector_type(8)))  float  v8f;

// ---------------------------------------------------------------------------
// CDNA5 async memory->LDS copy of 32 bytes (2 x b128), tracked by ASYNCcnt.
// offset:N is added to the LDS address and to the memory address alike,
// i.e. the immediate offset advances BOTH addresses.
// ---------------------------------------------------------------------------
__device__ __forceinline__ void async_copy32(const __bf16* src, __bf16* dstLds) {
    const unsigned lds = (unsigned)(size_t)(void*)dstLds;      // low 32b = LDS offset
    const unsigned long long ga = (unsigned long long)(size_t)src;
    asm volatile("global_load_async_to_lds_b128 %0, %1, off\n\t"
                 "global_load_async_to_lds_b128 %0, %1, off offset:16"
                 :: "v"(lds), "v"(ga) : "memory");
}

__device__ __forceinline__ void wait_async0() {
    asm volatile("s_wait_asynccnt 0x0" ::: "memory");
}

// ---------------------------------------------------------------------------
// Kernel 1: codebook f32 -> bf16, and per-code squared norms ||c||^2
// grid = K blocks, 256 threads (= D)
// ---------------------------------------------------------------------------
__global__ void __launch_bounds__(256)
vq_prep_kernel(const float* __restrict__ emb,
               __bf16* __restrict__ cb,
               float* __restrict__ csqr) {
    __shared__ float red[256];
    const int row = blockIdx.x;
    const int tid = threadIdx.x;
    const float v = emb[(size_t)row * DDIM + tid];
    cb[(size_t)row * DDIM + tid] = (__bf16)v;
    red[tid] = v * v;
    __syncthreads();
    #pragma unroll
    for (int s = 128; s > 0; s >>= 1) {
        if (tid < s) red[tid] += red[tid + s];
        __syncthreads();
    }
    if (tid == 0) csqr[row] = red[0];
}

// ---------------------------------------------------------------------------
// Kernel 2: distances + argmin via v_wmma_f32_16x16x32_bf16
// grid = 256 blocks x 256 threads (8 waves). Wave w of block b owns rows
// [(b*8+w)*16, +16). Workgroup stages 16-code B tiles into double-buffered
// LDS with GLOBAL_LOAD_ASYNC_TO_LDS_B128 issued one tile ahead.
// argmin_n(||x||^2 + ||c_n||^2 - 2 x.c_n) == argmin_n(||c_n||^2 - 2 x.c_n)
// ---------------------------------------------------------------------------
__global__ void __launch_bounds__(256)
vq_argmin_kernel(const float* __restrict__ x,     // [N, D] f32
                 const __bf16* __restrict__ cb,   // [K, D] bf16
                 const float* __restrict__ csqr,  // [K]
                 float* __restrict__ idx_out_f,   // [N] (float-coded indices)
                 int*   __restrict__ idx_out_i) { // [N] int32 for gather
    __shared__ __bf16 tileB[2][TILE_N * LDS_STRIDE];
    __shared__ float  csqrLds[KCODES];

    const int tid  = threadIdx.x;
    const int lane = tid & 31;
    const int wave = tid >> 5;
    const int n    = lane & 15;   // column (code) within tile / row within A-block
    const int hl   = lane >> 4;   // half-wave select
    const int rowBase = (blockIdx.x * 8 + wave) * 16;
    const int myRow   = rowBase + n;

    // stage ||c||^2 into LDS once
    for (int i = tid; i < KCODES; i += 256) csqrLds[i] = csqr[i];

    // --- Preload this wave's 16x256 A block in contiguous k order (k = k0 + 16*hl + s), same as B ---
    // lane<16 : elems 0..15 -> K = k0+0..15
    // lane>=16: elems 0..15 -> K = k0+16..31
    v16bf a[NKCH];
    {
        const float* xrow = x + (size_t)myRow * DDIM;
        #pragma unroll
        for (int kk = 0; kk < NKCH; ++kk) {
            const int k0 = kk * KCHUNK;
            const float4* p0 = reinterpret_cast<const float4*>(xrow + k0 + hl * 16);
            const float4* p1 = reinterpret_cast<const float4*>(xrow + k0 + hl * 16 + 8);
            const float4 f0 = p0[0], f1 = p0[1];
            const float4 f2 = p1[0], f3 = p1[1];
            a[kk][0]  = (__bf16)f0.x; a[kk][1]  = (__bf16)f0.y;
            a[kk][2]  = (__bf16)f0.z; a[kk][3]  = (__bf16)f0.w;
            a[kk][4]  = (__bf16)f1.x; a[kk][5]  = (__bf16)f1.y;
            a[kk][6]  = (__bf16)f1.z; a[kk][7]  = (__bf16)f1.w;
            a[kk][8]  = (__bf16)f2.x; a[kk][9]  = (__bf16)f2.y;
            a[kk][10] = (__bf16)f2.z; a[kk][11] = (__bf16)f2.w;
            a[kk][12] = (__bf16)f3.x; a[kk][13] = (__bf16)f3.y;
            a[kk][14] = (__bf16)f3.z; a[kk][15] = (__bf16)f3.w;
        }
    }

    float minv[8];
    int   mini[8];
    #pragma unroll
    for (int r = 0; r < 8; ++r) { minv[r] = 3.402823466e38f; mini[r] = 0; }

    // per-thread staging slot: 32B = 16 bf16
    const int ldRow   = tid >> 4;   // 0..15 : code row within tile
    const int ldChunk = tid & 15;   // 0..15 : 16-elem chunk along D
    __bf16* dst0 = &tileB[0][ldRow * LDS_STRIDE + ldChunk * 16];
    __bf16* dst1 = &tileB[1][ldRow * LDS_STRIDE + ldChunk * 16];
    const __bf16* srcBase = cb + (size_t)ldRow * DDIM + ldChunk * 16;

    // prologue: async-stage tile 0
    async_copy32(srcBase, dst0);
    wait_async0();
    __syncthreads();

    for (int t = 0; t < NTILES; ++t) {
        // issue async staging for tile t+1 into the other buffer (overlaps compute)
        if (t + 1 < NTILES)
            async_copy32(srcBase + (size_t)(t + 1) * TILE_N * DDIM,
                         (t & 1) ? dst0 : dst1);
        // prefetch tile t+2 toward near caches
        if (t + 2 < NTILES)
            __builtin_prefetch(srcBase + (size_t)(t + 2) * TILE_N * DDIM, 0, 3);

        // accumulate x . c over full D for this 16x16 tile
        const __bf16* bbase = &tileB[t & 1][n * LDS_STRIDE + hl * 16];
        v8f acc = {0.f, 0.f, 0.f, 0.f, 0.f, 0.f, 0.f, 0.f};
        v16bf bcur, bnxt;
        #pragma unroll
        for (int i = 0; i < 16; ++i) bcur[i] = bbase[i];
        #pragma unroll
        for (int kk = 0; kk < NKCH; ++kk) {
            if (kk + 1 < NKCH) {            // issue next chunk's LDS loads early
                const __bf16* bp = bbase + (kk + 1) * KCHUNK;
                #pragma unroll
                for (int i = 0; i < 16; ++i) bnxt[i] = bp[i];
            }
            acc = __builtin_amdgcn_wmma_f32_16x16x32_bf16(
                false, a[kk], false, bcur, (short)0, acc, false, false);
            if (kk + 1 < NKCH) bcur = bnxt;
        }

        // D layout: lane<16: (M=r, N=n); lane>=16: (M=8+r, N=n)
        const float cn    = csqrLds[t * TILE_N + n];
        const int   nGlob = t * TILE_N + n;
        #pragma unroll
        for (int r = 0; r < 8; ++r) {
            const float d = cn - 2.0f * acc[r];
            if (d < minv[r]) { minv[r] = d; mini[r] = nGlob; }
        }

        // tile t+1 fully landed in LDS across all waves before next iteration
        wait_async0();
        __syncthreads();
    }

    // cross-lane argmin over the 16 N-lanes (butterfly within each half-wave),
    // first-min (lowest-index) tie-break to match jnp.argmin
    #pragma unroll
    for (int m = 1; m <= 8; m <<= 1) {
        #pragma unroll
        for (int r = 0; r < 8; ++r) {
            const float ov = __shfl_xor(minv[r], m, 32);
            const int   oi = __shfl_xor(mini[r], m, 32);
            if (ov < minv[r] || (ov == minv[r] && oi < mini[r])) {
                minv[r] = ov; mini[r] = oi;
            }
        }
    }
    if (n == 0) {
        #pragma unroll
        for (int r = 0; r < 8; ++r) {
            const int row = rowBase + r + 8 * hl;
            idx_out_f[row] = (float)mini[r];
            idx_out_i[row] = mini[r];
        }
    }
}

// ---------------------------------------------------------------------------
// Kernel 3: gather z_q = z_q_bar = embedding[idx]; 4 rows per 256-thread block
// ---------------------------------------------------------------------------
__global__ void __launch_bounds__(256)
vq_gather_kernel(const float* __restrict__ emb,
                 const int* __restrict__ idx,
                 float* __restrict__ zq,
                 float* __restrict__ zqbar) {
    const int row  = blockIdx.x * 4 + (threadIdx.x >> 6);
    const int t    = threadIdx.x & 63;              // 64 x float4 = 256 floats
    const int code = idx[row];
    const float4 v = reinterpret_cast<const float4*>(emb)[(size_t)code * 64 + t];
    reinterpret_cast<float4*>(zq)[(size_t)row * 64 + t]    = v;
    reinterpret_cast<float4*>(zqbar)[(size_t)row * 64 + t] = v;
}

// ---------------------------------------------------------------------------
extern "C" void kernel_launch(void* const* d_in, const int* in_sizes, int n_in,
                              void* d_out, int out_size, void* d_ws, size_t ws_size,
                              hipStream_t stream) {
    const float* z_e_x     = (const float*)d_in[0];   // [B,T,D] f32
    const float* embedding = (const float*)d_in[1];   // [K,D]   f32

    float* out   = (float*)d_out;
    float* zq    = out;                                   // [N,D]
    float* zqbar = out + (size_t)N_ROWS * DDIM;           // [N,D]
    float* idxf  = out + 2 * (size_t)N_ROWS * DDIM;       // [N] (float-coded)

    char* ws = (char*)d_ws;
    __bf16* cb   = (__bf16*)(void*)ws;                    // 512 KB bf16 codebook
    float*  csqr = (float*)(void*)(ws + 512 * 1024);      //   4 KB ||c||^2
    int*    idxi = (int*)(void*)(ws + 516 * 1024);        // 128 KB int indices

    vq_prep_kernel<<<KCODES, 256, 0, stream>>>(embedding, cb, csqr);
    vq_argmin_kernel<<<N_ROWS / (8 * 16), 256, 0, stream>>>(z_e_x, cb, csqr, idxf, idxi);
    vq_gather_kernel<<<N_ROWS / 4, 256, 0, stream>>>(embedding, idxi, zq, zqbar);
}
